// ROLANDGNN_75651553951753
// MI455X (gfx1250) — matrix operand order repaired and checked by reading, not compiled, not run on hardware
//
#include <hip/hip_runtime.h>
#include <math.h>

typedef __attribute__((ext_vector_type(16))) __bf16 v16bf;
typedef __attribute__((ext_vector_type(8)))  __bf16 v8bf;
typedef __attribute__((ext_vector_type(8)))  float  v8f;
typedef __attribute__((ext_vector_type(4)))  int    v4i;

#define AS_GLOBAL __attribute__((address_space(1)))
#define AS_LDS    __attribute__((address_space(3)))

#if defined(__gfx1250__) && __has_builtin(__builtin_amdgcn_global_load_async_to_lds_b128)
#define USE_ASYNC_LDS 1
#else
#define USE_ASYNC_LDS 0
#endif

constexpr int kN  = 50000;   // nodes
constexpr int kIN = 128;     // input dim
constexpr int kH1 = 256;
constexpr int kH2 = 128;
constexpr int kC  = 128;     // C1 == C2
constexpr int kE  = 800000;  // edges

__device__ __forceinline__ float leaky(float v) { return v > 0.0f ? v : 0.01f * v; }
__device__ __forceinline__ float sigmoidf(float v) { return 1.0f / (1.0f + __expf(-v)); }

// ---------------------------------------------------------------- utility
__global__ void fill_zero_kernel(float* __restrict__ p, long n) {
    long i = (long)blockIdx.x * blockDim.x + threadIdx.x;
    if (i < n) p[i] = 0.0f;
}
__global__ void cvt_bf16_kernel(const float* __restrict__ s, __bf16* __restrict__ d, long n) {
    long i = (long)blockIdx.x * blockDim.x + threadIdx.x;
    if (i < n) d[i] = (__bf16)s[i];
}

// ---------------------------------------------------------------- degree / norm
__global__ void deg_init_kernel(float* __restrict__ deg) {
    int i = blockIdx.x * blockDim.x + threadIdx.x;
    if (i < kN) deg[i] = 1.0f;                       // self loop contributes 1
}
__global__ void deg_accum_kernel(const int* __restrict__ dst, float* __restrict__ deg) {
    int e = blockIdx.x * blockDim.x + threadIdx.x;
    if (e < kE) atomicAdd(&deg[dst[e]], 1.0f);
}
__global__ void deg_finish_kernel(const float* __restrict__ deg, float* __restrict__ dis) {
    int i = blockIdx.x * blockDim.x + threadIdx.x;
    if (i < kN) dis[i] = rsqrtf(deg[i]);             // deg >= 1 always
}

// ---------------------------------------------------------------- WMMA GEMM (bf16 in)
// C[M,Ndim] = act( A[M,K] @ W[Ndim,K]^T + bias ), A/W bf16, C f32 or bf16.
// Block tile 128x64, 8 waves (4x2), each wave a 32x32 patch = 2x2 WMMA 16x16x32 tiles.
// LDS staged with GLOBAL_LOAD_ASYNC_TO_LDS_B128 when available (tracked by ASYNCcnt).
#define BM 128
#define BN 64
#define LK 32
#define LPITCH (LK + 8)

__global__ __launch_bounds__(256) void gemm_bf16_kernel(
    const __bf16* __restrict__ A, const __bf16* __restrict__ W,
    const float* __restrict__ bias, float* __restrict__ Cf, __bf16* __restrict__ Cb,
    int M, int Ndim, int K, int act)
{
    __shared__ __bf16 lA[BM][LPITCH];
    __shared__ __bf16 lB[BN][LPITCH];

    const int tid  = threadIdx.x;
    const int lane = tid & 31;
    const int wave = tid >> 5;
    const int wm   = wave & 3;   // row sub-tile (x32)
    const int wn   = wave >> 2;  // col sub-tile (x32)

    const long rowBase = (long)blockIdx.x * BM;
    const int  colBase = blockIdx.y * BN;

    v8f acc[2][2] = {};

    const int r16 = lane & 15;
    const int hi  = lane >> 4;    // 0 or 1: k-halves per ISA VGPR layout

    const int browB = tid >> 2;          // B: 4 threads/row, 8 bf16 each
    const int bcoB  = (tid & 3) * 8;

    for (int k0 = 0; k0 < K; k0 += LK) {
        // ---- stage A tile (128 x 32 bf16): 512 x 16B segments, 2 per thread
        #pragma unroll
        for (int t = 0; t < 2; ++t) {
            int seg = tid * 2 + t;
            int row = seg >> 2;
            int co  = (seg & 3) * 8;
            long gr = rowBase + row;
#if USE_ASYNC_LDS
            if (gr < M) {
                __builtin_amdgcn_global_load_async_to_lds_b128(
                    (AS_GLOBAL v4i*)(A + gr * (long)K + k0 + co),
                    (AS_LDS v4i*)&lA[row][co], 0, 0);
            } else {
                v8bf z = {};
                *(v8bf*)&lA[row][co] = z;
            }
#else
            v8bf v = {};
            if (gr < M) v = *(const v8bf*)(A + gr * (long)K + k0 + co);
            *(v8bf*)&lA[row][co] = v;
#endif
        }
        // ---- stage B tile (64 x 32 bf16): 256 x 16B segments, 1 per thread
        {
#if USE_ASYNC_LDS
            __builtin_amdgcn_global_load_async_to_lds_b128(
                (AS_GLOBAL v4i*)(W + (long)(colBase + browB) * K + k0 + bcoB),
                (AS_LDS v4i*)&lB[browB][bcoB], 0, 0);
#else
            *(v8bf*)&lB[browB][bcoB] =
                *(const v8bf*)(W + (long)(colBase + browB) * K + k0 + bcoB);
#endif
        }
#if USE_ASYNC_LDS
  #if __has_builtin(__builtin_amdgcn_s_wait_asynccnt)
        __builtin_amdgcn_s_wait_asynccnt(0);
  #else
        asm volatile("s_wait_asynccnt 0x0" ::: "memory");
  #endif
#endif
        __syncthreads();

        // ---- fragments per ISA 16-bit layouts
        v16bf aF[2], bF[2];
        #pragma unroll
        for (int i = 0; i < 2; ++i) {
            int rr = wm * 32 + i * 16 + r16;
            // lanes 0-15: K = {0..7, 16..23}; lanes 16-31: K = {8..15, 24..31}
            v8bf lo  = *(const v8bf*)&lA[rr][hi * 8];
            v8bf hi8 = *(const v8bf*)&lA[rr][hi * 8 + 16];
            aF[i] = __builtin_shufflevector(lo, hi8,
                     0,1,2,3,4,5,6,7,8,9,10,11,12,13,14,15);
        }
        #pragma unroll
        for (int j = 0; j < 2; ++j) {
            int cc = wn * 32 + j * 16 + r16;
            // lanes 0-15: K = {0..7, 16..23}; lanes 16-31: K = {8..15, 24..31}
            v8bf lo  = *(const v8bf*)&lB[cc][hi * 8];
            v8bf hi8 = *(const v8bf*)&lB[cc][hi * 8 + 16];
            bF[j] = __builtin_shufflevector(lo, hi8,
                     0,1,2,3,4,5,6,7,8,9,10,11,12,13,14,15);
        }

        #pragma unroll
        for (int i = 0; i < 2; ++i)
            #pragma unroll
            for (int j = 0; j < 2; ++j)
                acc[i][j] = __builtin_amdgcn_wmma_f32_16x16x32_bf16(
                    false, aF[i], false, bF[j], (short)0, acc[i][j], false, false);

        __syncthreads();
    }

    // ---- epilogue: C/D layout: VGPR r, lanes 0-15 -> M=r, lanes 16-31 -> M=r+8
    #pragma unroll
    for (int i = 0; i < 2; ++i) {
        #pragma unroll
        for (int j = 0; j < 2; ++j) {
            #pragma unroll
            for (int r = 0; r < 8; ++r) {
                long row = rowBase + wm * 32 + i * 16 + hi * 8 + r;
                int  col = colBase + wn * 32 + j * 16 + r16;
                if (row < M) {
                    float v = acc[i][j][r];
                    if (bias) v += bias[col];
                    if (act == 1) v = leaky(v);
                    if (Cf) Cf[row * (long)Ndim + col] = v;
                    else    Cb[row * (long)Ndim + col] = (__bf16)v;
                }
            }
        }
    }
}

// ---------------------------------------------------------------- GCN scatter
// one wave per (edge or self loop): agg[dst] += m[src] * dis[src]*dis[dst]
__global__ __launch_bounds__(256) void gcn_scatter_kernel(
    const float* __restrict__ m, const int* __restrict__ src, const int* __restrict__ dst,
    const float* __restrict__ dis, float* __restrict__ agg)
{
    long edge = ((long)blockIdx.x * blockDim.x + threadIdx.x) >> 5;
    int  lane = threadIdx.x & 31;
    if (edge >= (long)kE + kN) return;
    int s, d;
    if (edge < kE) { s = src[edge]; d = dst[edge]; }
    else           { s = d = (int)(edge - kE); }
    float norm = dis[s] * dis[d];
    const float* mrow = m + (long)s * kC;
    float*       arow = agg + (long)d * kC;
    #pragma unroll
    for (int j = 0; j < kC / 32; ++j) {
        int c = lane + j * 32;
        atomicAdd(&arow[c], mrow[c] * norm);
    }
}

// ---------------------------------------------------------------- bias + leaky -> bf16
// (GCN activation is consumed only by the following GRU-gate GEMM)
__global__ void bias_leaky_bf16_kernel(const float* __restrict__ x, const float* __restrict__ b,
                                       __bf16* __restrict__ out, long total) {
    long i = (long)blockIdx.x * blockDim.x + threadIdx.x;
    if (i < total) {
        int c = (int)(i & (kC - 1));
        out[i] = (__bf16)leaky(x[i] + b[c]);
    }
}

// ---------------------------------------------------------------- GRU cell + L2 norm
// one wave per row (128 channels -> 4 per lane); optional bf16 copy for next GEMM
__global__ __launch_bounds__(256) void gru_norm_kernel(
    const float* __restrict__ gi, const float* __restrict__ gh,
    const float* __restrict__ hprev, float* __restrict__ out1, float* __restrict__ out2,
    __bf16* __restrict__ outb)
{
    long row  = ((long)blockIdx.x * blockDim.x + threadIdx.x) >> 5;
    int  lane = threadIdx.x & 31;
    if (row >= kN) return;
    const float* gir = gi + row * (long)(3 * kC);
    const float* ghr = gh + row * (long)(3 * kC);
    const float* hr  = hprev + row * (long)kC;

    float vals[kC / 32];
    float ss = 0.0f;
    #pragma unroll
    for (int j = 0; j < kC / 32; ++j) {
        int c = lane + j * 32;
        float i_r = gir[c], i_z = gir[kC + c], i_n = gir[2 * kC + c];
        float h_r = ghr[c], h_z = ghr[kC + c], h_n = ghr[2 * kC + c];
        float r = sigmoidf(i_r + h_r);
        float z = sigmoidf(i_z + h_z);
        float n = tanhf(i_n + r * h_n);
        float v = (1.0f - z) * n + z * hr[c];
        vals[j] = v;
        ss += v * v;
    }
    #pragma unroll
    for (int o = 16; o > 0; o >>= 1) ss += __shfl_xor(ss, o, 32);
    float rn = rsqrtf(ss);
    #pragma unroll
    for (int j = 0; j < kC / 32; ++j) {
        int c = lane + j * 32;
        float v = vals[j] * rn;
        out1[row * (long)kC + c] = v;
        if (out2) out2[row * (long)kC + c] = v;
        if (outb) outb[row * (long)kC + c] = (__bf16)v;
    }
}

// ---------------------------------------------------------------- launch
extern "C" void kernel_launch(void* const* d_in, const int* in_sizes, int n_in,
                              void* d_out, int out_size, void* d_ws, size_t ws_size,
                              hipStream_t stream) {
    const float* x         = (const float*)d_in[0];
    const int*   ei        = (const int*)  d_in[1];
    const float* prev1     = (const float*)d_in[2];
    const float* prev2     = (const float*)d_in[3];
    const float* w_pre1    = (const float*)d_in[4];
    const float* b_pre1    = (const float*)d_in[5];
    const float* w_pre2    = (const float*)d_in[6];
    const float* b_pre2    = (const float*)d_in[7];
    const float* w_conv1   = (const float*)d_in[8];
    const float* b_conv1   = (const float*)d_in[9];
    const float* w_conv2   = (const float*)d_in[10];
    const float* b_conv2   = (const float*)d_in[11];
    const float* gru1_wih  = (const float*)d_in[12];
    const float* gru1_whh  = (const float*)d_in[13];
    const float* gru1_bih  = (const float*)d_in[14];
    const float* gru1_bhh  = (const float*)d_in[15];
    const float* gru2_wih  = (const float*)d_in[16];
    const float* gru2_whh  = (const float*)d_in[17];
    const float* gru2_bih  = (const float*)d_in[18];
    const float* gru2_bhh  = (const float*)d_in[19];

    const int* esrc = ei;
    const int* edst = ei + kE;

    float* o        = (float*)d_out;
    float* out_h    = o;                                 // == emb2
    float* out_emb1 = o + (size_t)kN * kC;
    float* out_emb2 = o + (size_t)kN * kC * 2;

    // ---- workspace carve-up
    char* wp = (char*)d_ws;
    size_t n = kN;
    auto allocF = [&](size_t nf) { float* r = (float*)wp; wp += nf * sizeof(float); return r; };
    auto allocB = [&](size_t nb) { __bf16* r = (__bf16*)wp; wp += nb * sizeof(__bf16); return r; };

    float* deg = allocF(n);
    float* dis = allocF(n);
    float* m   = allocF(n * 128);       // conv message (f32, scatter input)
    float* agg = allocF(n * 128);       // scatter aggregate
    float* gi  = allocF(n * 384);
    float* gh  = allocF(n * 384);

    __bf16* xb     = allocB(n * 128);   // bf16 copies of GEMM inputs
    __bf16* prev1b = allocB(n * 128);
    __bf16* prev2b = allocB(n * 128);
    __bf16* h1b    = allocB(n * 256);
    __bf16* h2b    = allocB(n * 128);
    __bf16* gcnb   = allocB(n * 128);   // GCN activation (reused both layers)
    __bf16* emb1b  = allocB(n * 128);
    __bf16* wpre1b  = allocB((size_t)kH1 * kIN);
    __bf16* wpre2b  = allocB((size_t)kH2 * kH1);
    __bf16* wconv1b = allocB((size_t)kC * kH2);
    __bf16* wconv2b = allocB((size_t)kC * kC);
    __bf16* g1wihb  = allocB((size_t)3 * kC * kC);
    __bf16* g1whhb  = allocB((size_t)3 * kC * kC);
    __bf16* g2wihb  = allocB((size_t)3 * kC * kC);
    __bf16* g2whhb  = allocB((size_t)3 * kC * kC);

    const int T = 256;
    const int MT = (kN + BM - 1) / BM;                 // 391 row tiles
    const long aggTot = (long)kN * kC;
    const int aggBlocks = (int)((aggTot + T - 1) / T);
    const long scatLanes = ((long)kE + kN) * 32;
    const int scatBlocks = (int)((scatLanes + T - 1) / T);
    const int rowWaveBlocks = (kN * 32 + T - 1) / T;
    auto cvB = [&](long cnt) { return (int)((cnt + T - 1) / T); };

    // ---- degrees / symmetric norm
    deg_init_kernel<<<(kN + T - 1) / T, T, 0, stream>>>(deg);
    deg_accum_kernel<<<(kE + T - 1) / T, T, 0, stream>>>(edst, deg);
    deg_finish_kernel<<<(kN + T - 1) / T, T, 0, stream>>>(deg, dis);

    // ---- one-time bf16 conversions (inputs + weights)
    cvt_bf16_kernel<<<cvB((long)n * 128), T, 0, stream>>>(x,     xb,     (long)n * 128);
    cvt_bf16_kernel<<<cvB((long)n * 128), T, 0, stream>>>(prev1, prev1b, (long)n * 128);
    cvt_bf16_kernel<<<cvB((long)n * 128), T, 0, stream>>>(prev2, prev2b, (long)n * 128);
    cvt_bf16_kernel<<<cvB((long)kH1 * kIN), T, 0, stream>>>(w_pre1,  wpre1b,  (long)kH1 * kIN);
    cvt_bf16_kernel<<<cvB((long)kH2 * kH1), T, 0, stream>>>(w_pre2,  wpre2b,  (long)kH2 * kH1);
    cvt_bf16_kernel<<<cvB((long)kC * kH2),  T, 0, stream>>>(w_conv1, wconv1b, (long)kC * kH2);
    cvt_bf16_kernel<<<cvB((long)kC * kC),   T, 0, stream>>>(w_conv2, wconv2b, (long)kC * kC);
    cvt_bf16_kernel<<<cvB((long)3 * kC * kC), T, 0, stream>>>(gru1_wih, g1wihb, (long)3 * kC * kC);
    cvt_bf16_kernel<<<cvB((long)3 * kC * kC), T, 0, stream>>>(gru1_whh, g1whhb, (long)3 * kC * kC);
    cvt_bf16_kernel<<<cvB((long)3 * kC * kC), T, 0, stream>>>(gru2_wih, g2wihb, (long)3 * kC * kC);
    cvt_bf16_kernel<<<cvB((long)3 * kC * kC), T, 0, stream>>>(gru2_whh, g2whhb, (long)3 * kC * kC);

    // ---- MLP preprocess (bf16 out: consumed only by next GEMM)
    gemm_bf16_kernel<<<dim3(MT, kH1 / BN), T, 0, stream>>>(xb,  wpre1b, b_pre1, nullptr, h1b, kN, kH1, kIN, 1);
    gemm_bf16_kernel<<<dim3(MT, kH2 / BN), T, 0, stream>>>(h1b, wpre2b, b_pre2, nullptr, h2b, kN, kH2, kH1, 1);

    // ---- layer 1: GCN -> leaky -> GRU -> L2norm
    gemm_bf16_kernel<<<dim3(MT, kC / BN), T, 0, stream>>>(h2b, wconv1b, nullptr, m, nullptr, kN, kC, kH2, 0);
    fill_zero_kernel<<<aggBlocks, T, 0, stream>>>(agg, aggTot);
    gcn_scatter_kernel<<<scatBlocks, T, 0, stream>>>(m, esrc, edst, dis, agg);
    bias_leaky_bf16_kernel<<<aggBlocks, T, 0, stream>>>(agg, b_conv1, gcnb, aggTot);
    gemm_bf16_kernel<<<dim3(MT, 3 * kC / BN), T, 0, stream>>>(gcnb,   g1wihb, gru1_bih, gi, nullptr, kN, 3 * kC, kC, 0);
    gemm_bf16_kernel<<<dim3(MT, 3 * kC / BN), T, 0, stream>>>(prev1b, g1whhb, gru1_bhh, gh, nullptr, kN, 3 * kC, kC, 0);
    gru_norm_kernel<<<rowWaveBlocks, T, 0, stream>>>(gi, gh, prev1, out_emb1, nullptr, emb1b);

    // ---- layer 2: GCN -> leaky -> GRU -> L2norm
    gemm_bf16_kernel<<<dim3(MT, kC / BN), T, 0, stream>>>(emb1b, wconv2b, nullptr, m, nullptr, kN, kC, kC, 0);
    fill_zero_kernel<<<aggBlocks, T, 0, stream>>>(agg, aggTot);
    gcn_scatter_kernel<<<scatBlocks, T, 0, stream>>>(m, esrc, edst, dis, agg);
    bias_leaky_bf16_kernel<<<aggBlocks, T, 0, stream>>>(agg, b_conv2, gcnb, aggTot);
    gemm_bf16_kernel<<<dim3(MT, 3 * kC / BN), T, 0, stream>>>(gcnb,   g2wihb, gru2_bih, gi, nullptr, kN, 3 * kC, kC, 0);
    gemm_bf16_kernel<<<dim3(MT, 3 * kC / BN), T, 0, stream>>>(prev2b, g2whhb, gru2_bhh, gh, nullptr, kN, 3 * kC, kC, 0);
    gru_norm_kernel<<<rowWaveBlocks, T, 0, stream>>>(gi, gh, prev2, out_h, out_emb2, nullptr);
}
